// Encoder_7361573945979
// MI455X (gfx1250) — matrix operand order repaired and checked by reading, not compiled, not run on hardware
//
#include <hip/hip_runtime.h>
#include <hip/hip_bf16.h>

typedef _Float16 half_t;
typedef __attribute__((ext_vector_type(16))) _Float16 v16h;
typedef __attribute__((ext_vector_type(8)))  float    v8f;
typedef __attribute__((ext_vector_type(4)))  _Float16 v4h;
typedef int v4i __attribute__((vector_size(16)));   // matches builtin's V4i param

struct alignas(16) U4 { unsigned int x, y, z, w; };

union AFrag { v16h v; U4 q[2]; };
union BFrag { v16h v; U4 q[2]; };

// gfx1250 async global->LDS path (guarded; falls back to sync copies)
#if defined(__has_builtin)
#if __has_builtin(__builtin_amdgcn_global_load_async_to_lds_b128) && \
    __has_builtin(__builtin_amdgcn_s_wait_asynccnt)
#define USE_ASYNC 1
#endif
#endif
#ifndef USE_ASYNC
#define USE_ASYNC 0
#endif

#define GAS __attribute__((address_space(1)))
#define LAS __attribute__((address_space(3)))

// ---------------- conversion kernels ----------------

__global__ __launch_bounds__(256) void cvt_f32_to_f16_x4(const float* __restrict__ in,
                                                         half_t* __restrict__ out) {
    size_t i = (size_t)blockIdx.x * blockDim.x + threadIdx.x;
    const float4 f = ((const float4*)in)[i];
    v4h h;
    h.x = (half_t)f.x; h.y = (half_t)f.y; h.z = (half_t)f.z; h.w = (half_t)f.w;
    ((v4h*)out)[i] = h;
}

// in: [K][N] f32 (batched on z), out: [N][K] f16
__global__ __launch_bounds__(256) void transpose_cvt(const float* __restrict__ in,
                                                     half_t* __restrict__ out,
                                                     int K, int N) {
    __shared__ float tile[32][33];
    const size_t matOff = (size_t)blockIdx.z * (size_t)K * (size_t)N;
    const float* src = in + matOff;
    half_t* dst = out + matOff;
    const int n0 = blockIdx.x * 32, k0 = blockIdx.y * 32;
    const int tx = threadIdx.x & 31, ty = threadIdx.x >> 5; // 32 x 8
#pragma unroll
    for (int i = 0; i < 32; i += 8)
        tile[ty + i][tx] = src[(size_t)(k0 + ty + i) * N + (n0 + tx)];
    __syncthreads();
#pragma unroll
    for (int i = 0; i < 32; i += 8)
        dst[(size_t)(n0 + ty + i) * K + (k0 + tx)] = (half_t)tile[tx][ty + i];
}

// ---------------- routing ----------------

__global__ __launch_bounds__(256) void route_rows(const int* __restrict__ k,
                                                  int* __restrict__ counts,
                                                  int* __restrict__ row_ids, int Bn) {
    int i = blockIdx.x * 256 + threadIdx.x;
    if (i < Bn) {
        int e = k[i];
        int p = atomicAdd(&counts[e], 1);
        row_ids[(size_t)e * Bn + p] = i;
    }
}

// Pad each expert's list up to a multiple of BM by duplicating its first row.
// Duplicated rows recompute identical outputs -> writes are benign & deterministic.
__global__ __launch_bounds__(256) void pad_rows(const int* __restrict__ counts,
                                                int* __restrict__ row_ids,
                                                int Bn, int tileM) {
    const int e = blockIdx.x;
    const int cnt = counts[e];
    if (cnt == 0) return;
    const int padded = (cnt + tileM - 1) & ~(tileM - 1);
    const int fill = row_ids[(size_t)e * Bn];
    for (int i = cnt + threadIdx.x; i < padded; i += 256)
        row_ids[(size_t)e * Bn + i] = fill;
}

// ---------------- WMMA GEMM ----------------
// C[M,N] = act(A[M,K] @ Bt[N,K]^T + bias)
// GATHER: A rows from (padded) row_ids[e]; blockIdx.z = expert
// Block tile 128x256, 8 waves as 2(M) x 4(N), each wave computes 64x64.

#define BM 128
#define BN 256
#define BKK 64
#define LDST 72   // padded LDS row stride (halfs): 36 banks -> conflict-free frag reads
#define ACHUNK (BM * BKK / 8)   // 1024 16B chunks  (4 / thread)
#define BCHUNK (BN * BKK / 8)   // 2048 16B chunks  (8 / thread)

template <bool GATHER, bool RELU, bool OUT16>
__global__ __launch_bounds__(256) void gemm_wmma(const half_t* __restrict__ A,
                                                 const half_t* __restrict__ Bt,
                                                 const float* __restrict__ bias,
                                                 void* __restrict__ outp,
                                                 const int* __restrict__ row_ids,
                                                 const int* __restrict__ counts,
                                                 int M, int N, int K) {
    __shared__ half_t As[2][BM * LDST];
    __shared__ half_t Bs[2][BN * LDST];
    __shared__ int sidx[BM];

    const int tid  = threadIdx.x;
    const int lane = tid & 31;
    const int wave = tid >> 5;

    const int mbase = blockIdx.y * BM;
    const int nbase = blockIdx.x * BN;

    const half_t* Bmat = Bt;
    const float*  bvec = bias;
    if (GATHER) {
        const int e = blockIdx.z;
        const int cnt = counts[e];
        if (mbase >= cnt) return;                 // uniform early-exit
        const int* idxp = row_ids + (size_t)e * M + mbase;
        Bmat = Bt + (size_t)e * (size_t)N * (size_t)K;
        bvec = bias + (size_t)e * N;
        if (tid < BM) sidx[tid] = idxp[tid];      // padded -> always valid
        __syncthreads();
    }

    // issue one K-tile's global->LDS transfers (async on gfx1250, sync fallback)
    auto issueTile = [&](int bufi, int k0) {
        half_t* at = &As[bufi][0];
        half_t* bt = &Bs[bufi][0];
#pragma unroll
        for (int i = tid; i < ACHUNK; i += 256) {
            const int r = i >> 3, c = i & 7;
            const int grow = GATHER ? sidx[r] : (mbase + r);
            const half_t* gp = &A[(size_t)grow * K + k0 + c * 8];
            half_t* lp = &at[r * LDST + c * 8];
#if USE_ASYNC
            __builtin_amdgcn_global_load_async_to_lds_b128(
                (GAS v4i*)gp, (LAS v4i*)lp, 0, 0);
#else
            *(U4*)lp = *(const U4*)gp;
#endif
        }
#pragma unroll
        for (int i = tid; i < BCHUNK; i += 256) {
            const int r = i >> 3, c = i & 7;
            const half_t* gp = &Bmat[(size_t)(nbase + r) * K + k0 + c * 8];
            half_t* lp = &bt[r * LDST + c * 8];
#if USE_ASYNC
            __builtin_amdgcn_global_load_async_to_lds_b128(
                (GAS v4i*)gp, (LAS v4i*)lp, 0, 0);
#else
            *(U4*)lp = *(const U4*)gp;
#endif
        }
    };

    // wave tiling: 2 waves along M (64 rows), 4 waves along N (64 cols)
    const int mrow = (wave & 1) * 64;
    const int ncol = (wave >> 1) * 64;
    const int lm = lane & 15;
    const int lh = lane >> 4;

    v8f acc[4][4] = {};

    const int ktiles = K / BKK;
    issueTile(0, 0);
    for (int kt = 0; kt < ktiles; ++kt) {
        const int cur = kt & 1;
        const bool hasNext = (kt + 1) < ktiles;
        if (hasNext) issueTile(cur ^ 1, (kt + 1) * BKK);
#if USE_ASYNC
        // 12 async instrs per wave per tile; <=12 pending means current tile landed
        if (hasNext) __builtin_amdgcn_s_wait_asynccnt(12);
        else         __builtin_amdgcn_s_wait_asynccnt(0);
#endif
        __syncthreads();

        const half_t* Ab = &As[cur][0];
        const half_t* Bb = &Bs[cur][0];
#pragma unroll
        for (int ki = 0; ki < BKK / 32; ++ki) {
            const int kk = ki * 32;
            AFrag a[4];
            BFrag b[4];
#pragma unroll
            for (int mi = 0; mi < 4; ++mi) {
                const half_t* p = &Ab[(mrow + mi * 16 + lm) * LDST + kk + lh * 8];
                a[mi].q[0] = *(const U4*)p;          // K = kk + lh*8 .. +7
                a[mi].q[1] = *(const U4*)(p + 16);   // K = kk + 16 + lh*8 .. +7
            }
#pragma unroll
            for (int ni = 0; ni < 4; ++ni) {
                const half_t* p = &Bb[(ncol + ni * 16 + lm) * LDST + kk + lh * 8];
                b[ni].q[0] = *(const U4*)p;          // K = kk + lh*8 .. +7
                b[ni].q[1] = *(const U4*)(p + 16);   // K = kk + 16 + lh*8 .. +7
            }
#pragma unroll
            for (int mi = 0; mi < 4; ++mi)
#pragma unroll
                for (int ni = 0; ni < 4; ++ni)
                    acc[mi][ni] = __builtin_amdgcn_wmma_f32_16x16x32_f16(
                        false, a[mi].v, false, b[ni].v, (short)0, acc[mi][ni],
                        false, false);
        }
        __syncthreads();
    }

    // ---- epilogue: bias (+ReLU) and store ----
#pragma unroll
    for (int mi = 0; mi < 4; ++mi)
#pragma unroll
        for (int ni = 0; ni < 4; ++ni) {
            const int col = nbase + ncol + ni * 16 + lm;
            const float bv = bvec[col];
#pragma unroll
            for (int j = 0; j < 8; ++j) {
                const int lr = mrow + mi * 16 + lh * 8 + j;   // local row 0..127
                float v = acc[mi][ni][j] + bv;
                if (RELU) v = v > 0.0f ? v : 0.0f;
                const int grow = GATHER ? sidx[lr] : (mbase + lr);
                if (OUT16)
                    ((half_t*)outp)[(size_t)grow * N + col] = (half_t)v;
                else
                    ((float*)outp)[(size_t)grow * N + col] = v;
            }
        }
}

// ---------------- launcher ----------------

extern "C" void kernel_launch(void* const* d_in, const int* in_sizes, int n_in,
                              void* d_out, int out_size, void* d_ws, size_t ws_size,
                              hipStream_t stream) {
    constexpr int Bn = 16384, DIN = 2048, DC = 1024, H = 4096, DO = 1024, E = 4;

    const float* x     = (const float*)d_in[0];
    const int*   kidx  = (const int*)d_in[1];
    const float* W_exp = (const float*)d_in[2];
    const float* b_exp = (const float*)d_in[3];
    const float* W1    = (const float*)d_in[4];
    const float* b1    = (const float*)d_in[5];
    const float* W2    = (const float*)d_in[6];
    const float* b2    = (const float*)d_in[7];
    const float* W3    = (const float*)d_in[8];
    const float* b3    = (const float*)d_in[9];

    char* ws = (char*)d_ws;
    size_t off = 0;
    auto alloc = [&](size_t bytes) -> void* {
        void* p = ws + off;
        off = (off + bytes + 255) & ~(size_t)255;
        return p;
    };
    half_t* xh       = (half_t*)alloc((size_t)Bn * DIN * 2);
    half_t* WexpT    = (half_t*)alloc((size_t)E * DC * DIN * 2);
    half_t* W1T      = (half_t*)alloc((size_t)H * DC * 2);
    half_t* W2T      = (half_t*)alloc((size_t)H * H * 2);
    half_t* W3T      = (half_t*)alloc((size_t)DO * H * 2);
    half_t* alignedh = (half_t*)alloc((size_t)Bn * DC * 2);
    half_t* h1       = (half_t*)alloc((size_t)Bn * H * 2);
    half_t* h2       = (half_t*)alloc((size_t)Bn * H * 2);
    int*    counts   = (int*)alloc(E * sizeof(int));
    int*    row_ids  = (int*)alloc((size_t)E * Bn * sizeof(int));

    // 1) convert x to f16
    cvt_f32_to_f16_x4<<<((size_t)Bn * DIN / 4) / 256, 256, 0, stream>>>(x, xh);

    // 2) convert + transpose weights to f16 [N][K]
    transpose_cvt<<<dim3(DC / 32, DIN / 32, E), 256, 0, stream>>>(W_exp, WexpT, DIN, DC);
    transpose_cvt<<<dim3(H  / 32, DC  / 32, 1), 256, 0, stream>>>(W1, W1T, DC, H);
    transpose_cvt<<<dim3(H  / 32, H   / 32, 1), 256, 0, stream>>>(W2, W2T, H, H);
    transpose_cvt<<<dim3(DO / 32, H   / 32, 1), 256, 0, stream>>>(W3, W3T, H, DO);

    // 3) route rows to experts + pad lists to tile multiples
    (void)hipMemsetAsync(counts, 0, E * sizeof(int), stream);
    route_rows<<<Bn / 256, 256, 0, stream>>>(kidx, counts, row_ids, Bn);
    pad_rows<<<E, 256, 0, stream>>>(counts, row_ids, Bn, BM);

    // 4) gathered expert GEMM: aligned = x @ W_exp[k] + b_exp[k]   (f16 out)
    gemm_wmma<true, false, true><<<dim3(DC / BN, Bn / BM, E), 256, 0, stream>>>(
        xh, WexpT, b_exp, alignedh, row_ids, counts, Bn, DC, DIN);

    // 5) h1 = relu(aligned @ W1 + b1)
    gemm_wmma<false, true, true><<<dim3(H / BN, Bn / BM), 256, 0, stream>>>(
        alignedh, W1T, b1, h1, nullptr, nullptr, Bn, H, DC);

    // 6) h2 = relu(h1 @ W2 + b2)
    gemm_wmma<false, true, true><<<dim3(H / BN, Bn / BM), 256, 0, stream>>>(
        h1, W2T, b2, h2, nullptr, nullptr, Bn, H, H);

    // 7) out = h2 @ W3 + b3   (f32 out)
    gemm_wmma<false, false, false><<<dim3(DO / BN, Bn / BM), 256, 0, stream>>>(
        h2, W3T, b3, d_out, nullptr, nullptr, Bn, DO, H);
}
